// EventAugmentedLSTMCell_18193481465930
// MI455X (gfx1250) — matrix operand order repaired and checked by reading, not compiled, not run on hardware
//
#include <hip/hip_runtime.h>
#include <hip/hip_bf16.h>
#include <math.h>

typedef __attribute__((ext_vector_type(16))) __bf16 v16bf;
typedef __attribute__((ext_vector_type(8)))  float  v8f;

#define B_ROWS 16384
#define I_DIM  256
#define H_DIM  256
#define S_SLOTS 32
#define KCAT   768      // [x (256) ; h_mem (256) ; h_lstm (256)]
#define NGATES 1024     // 4*H

union FragB16 {
  v16bf v;
  uint4 q[2];
  unsigned short u[16];
};

__device__ __forceinline__ unsigned short f2bf(float f) {
  unsigned int u = __float_as_uint(f);
  unsigned int r = u + 0x7FFFu + ((u >> 16) & 1u);   // round-to-nearest-even
  return (unsigned short)(r >> 16);
}

// A fragment: 16x32 bf16, row-major source with leading dim lda.
// lanes 0-15: K = kk+0..7 (elems 0..7), kk+16..23 (elems 8..15)
// lanes 16-31: K = kk+8..15, kk+24..31
__device__ __forceinline__ void load_a(FragB16& f, const unsigned short* __restrict__ A,
                                       int row, int lda, int kk, int lane) {
  int kb = kk + ((lane & 16) ? 8 : 0);
  const unsigned short* p = A + (size_t)row * lda + kb;
  f.q[0] = *(const uint4*)(p);
  f.q[1] = *(const uint4*)(p + 16);
}

// B fragment: 32x16 bf16 from transposed weights Wt (N rows x K cols).
// lane = column N; lanes 0-15 hold K=kk+0..7, kk+16..23; lanes 16-31 hold K=kk+8..15, kk+24..31
__device__ __forceinline__ void load_b(FragB16& f, const unsigned short* __restrict__ Wt,
                                       int n, int K, int kk, int lane) {
  int ks = kk + ((lane & 16) ? 8 : 0);
  const unsigned short* p = Wt + (size_t)n * K + ks;
  f.q[0] = *(const uint4*)(p);
  f.q[1] = *(const uint4*)(p + 16);
}

// ---------- event gate: e = sigmoid(x @ W_ev + b_ev); mask; new_ptr ----------
__global__ void k_event(const float* __restrict__ x, const float* __restrict__ Wev,
                        const float* __restrict__ bev, const int* __restrict__ ptr,
                        int* __restrict__ mask, float* __restrict__ newptr_out) {
  int wave = threadIdx.x >> 5;
  int lane = threadIdx.x & 31;
  int row = blockIdx.x * 8 + wave;
  float p = 0.f;
#pragma unroll
  for (int j = 0; j < 8; ++j) {
    int k = lane + j * 32;
    p += x[(size_t)row * I_DIM + k] * Wev[k];
  }
#pragma unroll
  for (int off = 16; off > 0; off >>= 1) p += __shfl_xor(p, off, 32);
  if (lane == 0) {
    float z = p + bev[0];
    float e = 1.f / (1.f + __expf(-z));
    int m = (e > 0.85f) ? 1 : 0;
    mask[row] = m;
    newptr_out[row] = (float)((ptr[row] + m) & (S_SLOTS - 1));
  }
}

// ---------- build bf16 A-panel: Acat = [x ; (h_mem later) ; h_lstm] ----------
__global__ void k_acat(const float* __restrict__ x, const float* __restrict__ h,
                       unsigned short* __restrict__ Acat) {
  int idx = blockIdx.x * 256 + threadIdx.x;
  int b = idx >> 8, k = idx & 255;
  Acat[(size_t)b * KCAT + k]       = f2bf(x[idx]);
  Acat[(size_t)b * KCAT + 512 + k] = f2bf(h[idx]);
}

// ---------- transpose weights to bf16 (N rows x K cols, K contiguous) ----------
__global__ void k_wt(const float* __restrict__ Wval, const float* __restrict__ Wsp,
                     const float* __restrict__ Wih, const float* __restrict__ Whh,
                     unsigned short* __restrict__ WtV, unsigned short* __restrict__ WtS,
                     unsigned short* __restrict__ WtC) {
  int idx = blockIdx.x * 256 + threadIdx.x;
  if (idx < I_DIM * I_DIM) {
    int n = idx >> 8, k = idx & 255;
    WtV[idx] = f2bf(Wval[(size_t)k * I_DIM + n]);
    WtS[idx] = f2bf(Wsp[(size_t)k * H_DIM + n]);
  }
  if (idx < NGATES * KCAT) {
    int n = idx / KCAT, k = idx % KCAT;
    float s = (k < 512) ? Wih[(size_t)k * NGATES + n] : Whh[(size_t)(k - 512) * NGATES + n];
    WtC[idx] = f2bf(s);
  }
}

// ---------- GEMM: v = x @ W_val + b_val ----------
__global__ void k_gemm_v(const unsigned short* __restrict__ Acat,
                         const unsigned short* __restrict__ Wt,
                         const float* __restrict__ bias, float* __restrict__ out) {
  int wave = threadIdx.x >> 5, lane = threadIdx.x & 31;
  int half = (lane >> 4) & 1, col16 = lane & 15;
  int m0 = blockIdx.y * 128 + wave * 16, n0 = blockIdx.x * 16;
  v8f acc = {};
  for (int kk = 0; kk < I_DIM; kk += 32) {
    FragB16 a, b;
    load_a(a, Acat, m0 + col16, KCAT, kk, lane);
    load_b(b, Wt, n0 + col16, I_DIM, kk, lane);
    acc = __builtin_amdgcn_wmma_f32_16x16x32_bf16(false, a.v, false, b.v, (short)0, acc, false, false);
  }
  int col = n0 + col16;
  float bb = bias[col];
#pragma unroll
  for (int t = 0; t < 8; ++t) {
    int row = m0 + t + half * 8;
    out[(size_t)row * I_DIM + col] = acc[t] + bb;
  }
}

// ---------- softmax(slot_weights) and posdot[i] = sum_s w_s * pos_emb[s,i] ----------
__global__ void k_softpos(const float* __restrict__ sw_in, const float* __restrict__ pos,
                          float* __restrict__ w_out, float* __restrict__ posdot) {
  __shared__ float sw[S_SLOTS];
  int tid = threadIdx.x;
  if (tid == 0) {
    float mx = sw_in[0];
    for (int s = 1; s < S_SLOTS; ++s) mx = fmaxf(mx, sw_in[s]);
    float sum = 0.f;
    for (int s = 0; s < S_SLOTS; ++s) { float e = __expf(sw_in[s] - mx); sw[s] = e; sum += e; }
    float inv = 1.f / sum;
    for (int s = 0; s < S_SLOTS; ++s) { sw[s] *= inv; w_out[s] = sw[s]; }
  }
  __syncthreads();
  float acc = 0.f;
  for (int s = 0; s < S_SLOTS; ++s) acc += sw[s] * pos[s * I_DIM + tid];
  posdot[tid] = acc;
}

// ---------- single-pass slots stream: scatter-write + weighted pool (memory-bound core) ----------
__global__ void k_slots(const float* __restrict__ slots, const float* __restrict__ v,
                        const int* __restrict__ ptr, const int* __restrict__ mask,
                        const float* __restrict__ w, const float* __restrict__ posdot,
                        float* __restrict__ newslots, unsigned short* __restrict__ poolbf) {
  __shared__ float sw[S_SLOTS];
  int b = blockIdx.x, i = threadIdx.x;
  if (i < S_SLOTS) sw[i] = w[i];
  __syncthreads();
  int p = ptr[b] & (S_SLOTS - 1);
  int m = mask[b];
  float vv = v[(size_t)b * I_DIM + i];
  float acc = posdot[i];
  size_t base = (size_t)b * S_SLOTS * I_DIM + i;
#pragma unroll 4
  for (int s = 0; s < S_SLOTS; ++s) {
    float val = slots[base + (size_t)s * I_DIM];
    if (m && (s == p)) val = vv;
    newslots[base + (size_t)s * I_DIM] = val;
    acc += sw[s] * val;
  }
  poolbf[(size_t)b * I_DIM + i] = f2bf(acc);
}

// ---------- GEMM: h_mem = pooled @ W_sp + b_sp (also deposits bf16 into A-panel) ----------
__global__ void k_gemm_hmem(const unsigned short* __restrict__ poolbf,
                            const unsigned short* __restrict__ Wt,
                            const float* __restrict__ bias, float* __restrict__ hmem_out,
                            unsigned short* __restrict__ Acat) {
  int wave = threadIdx.x >> 5, lane = threadIdx.x & 31;
  int half = (lane >> 4) & 1, col16 = lane & 15;
  int m0 = blockIdx.y * 128 + wave * 16, n0 = blockIdx.x * 16;
  v8f acc = {};
  for (int kk = 0; kk < I_DIM; kk += 32) {
    FragB16 a, b;
    load_a(a, poolbf, m0 + col16, I_DIM, kk, lane);
    load_b(b, Wt, n0 + col16, I_DIM, kk, lane);
    acc = __builtin_amdgcn_wmma_f32_16x16x32_bf16(false, a.v, false, b.v, (short)0, acc, false, false);
  }
  int col = n0 + col16;
  float bb = bias[col];
#pragma unroll
  for (int t = 0; t < 8; ++t) {
    int row = m0 + t + half * 8;
    float val = acc[t] + bb;
    hmem_out[(size_t)row * H_DIM + col] = val;
    Acat[(size_t)row * KCAT + I_DIM + col] = f2bf(val);
  }
}

// ---------- fused gates GEMM (4 col-tiles per wave) + LSTM pointwise epilogue ----------
__global__ void k_gemm_lstm(const unsigned short* __restrict__ Acat,
                            const unsigned short* __restrict__ WtC,
                            const float* __restrict__ bih, const float* __restrict__ c_old,
                            float* __restrict__ h_new, float* __restrict__ c_new) {
  int wave = threadIdx.x >> 5, lane = threadIdx.x & 31;
  int half = (lane >> 4) & 1, col16 = lane & 15;
  int m0 = blockIdx.y * 128 + wave * 16, n0 = blockIdx.x * 16;
  v8f ai = {}, af = {}, ag = {}, ao = {};
  for (int kk = 0; kk < KCAT; kk += 32) {
    FragB16 a;
    load_a(a, Acat, m0 + col16, KCAT, kk, lane);
    FragB16 b0, b1, b2, b3;
    load_b(b0, WtC, n0 + col16,       KCAT, kk, lane);
    load_b(b1, WtC, n0 + col16 + 256, KCAT, kk, lane);
    load_b(b2, WtC, n0 + col16 + 512, KCAT, kk, lane);
    load_b(b3, WtC, n0 + col16 + 768, KCAT, kk, lane);
    ai = __builtin_amdgcn_wmma_f32_16x16x32_bf16(false, a.v, false, b0.v, (short)0, ai, false, false);
    af = __builtin_amdgcn_wmma_f32_16x16x32_bf16(false, a.v, false, b1.v, (short)0, af, false, false);
    ag = __builtin_amdgcn_wmma_f32_16x16x32_bf16(false, a.v, false, b2.v, (short)0, ag, false, false);
    ao = __builtin_amdgcn_wmma_f32_16x16x32_bf16(false, a.v, false, b3.v, (short)0, ao, false, false);
  }
  int col = n0 + col16;
  float bi = bih[col], bf_ = bih[col + 256], bg = bih[col + 512], bo = bih[col + 768];
#pragma unroll
  for (int t = 0; t < 8; ++t) {
    int row = m0 + t + half * 8;
    float gi = ai[t] + bi, gf = af[t] + bf_, gg = ag[t] + bg, go = ao[t] + bo;
    float co = c_old[(size_t)row * H_DIM + col];
    float si = 1.f / (1.f + __expf(-gi));
    float sf = 1.f / (1.f + __expf(-gf));
    float so = 1.f / (1.f + __expf(-go));
    float cn = sf * co + si * tanhf(gg);
    float hn = so * tanhf(cn);
    c_new[(size_t)row * H_DIM + col] = cn;
    h_new[(size_t)row * H_DIM + col] = hn;
  }
}

extern "C" void kernel_launch(void* const* d_in, const int* in_sizes, int n_in,
                              void* d_out, int out_size, void* d_ws, size_t ws_size,
                              hipStream_t stream) {
  const float* x     = (const float*)d_in[0];
  const float* hl    = (const float*)d_in[1];
  const float* cl    = (const float*)d_in[2];
  /* d_in[3] h_mem_prev is unused by the reference */
  const float* slots = (const float*)d_in[4];
  const int*   ptr   = (const int*)d_in[5];
  const float* Wval  = (const float*)d_in[6];
  const float* bval  = (const float*)d_in[7];
  const float* Wev   = (const float*)d_in[8];
  const float* bev   = (const float*)d_in[9];
  const float* pos   = (const float*)d_in[10];
  const float* swts  = (const float*)d_in[11];
  const float* Wsp   = (const float*)d_in[12];
  const float* bsp   = (const float*)d_in[13];
  const float* Wih   = (const float*)d_in[14];
  const float* bih   = (const float*)d_in[15];
  const float* Whh   = (const float*)d_in[16];

  float* out = (float*)d_out;
  const size_t BH = (size_t)B_ROWS * H_DIM;
  float* h_new  = out;
  float* c_new  = out + BH;
  float* hmem   = out + 2 * BH;
  float* nslots = out + 3 * BH;
  float* nptr   = out + 3 * BH + (size_t)B_ROWS * S_SLOTS * I_DIM;

  char* ws = (char*)d_ws;
  unsigned short* Acat   = (unsigned short*)(ws + 0);          // 25,165,824 B
  float*          vbuf   = (float*)(ws + 25165824);            // 16,777,216 B
  unsigned short* poolbf = (unsigned short*)(ws + 41943040);   //  8,388,608 B
  int*            mask   = (int*)(ws + 50331648);              //     65,536 B
  float*          wsm    = (float*)(ws + 50397184);            //        128 B
  float*          posdot = (float*)(ws + 50397312);            //      1,024 B
  unsigned short* WtV    = (unsigned short*)(ws + 50398336);   //    131,072 B
  unsigned short* WtS    = (unsigned short*)(ws + 50529408);   //    131,072 B
  unsigned short* WtC    = (unsigned short*)(ws + 50660480);   //  1,572,864 B

  k_event<<<B_ROWS / 8, 256, 0, stream>>>(x, Wev, bev, ptr, mask, nptr);
  k_acat<<<(B_ROWS * I_DIM) / 256, 256, 0, stream>>>(x, hl, Acat);
  k_wt<<<(NGATES * KCAT) / 256, 256, 0, stream>>>(Wval, Wsp, Wih, Whh, WtV, WtS, WtC);
  k_gemm_v<<<dim3(I_DIM / 16, B_ROWS / 128), 256, 0, stream>>>(Acat, WtV, bval, vbuf);
  k_softpos<<<1, 256, 0, stream>>>(swts, pos, wsm, posdot);
  k_slots<<<B_ROWS, 256, 0, stream>>>(slots, vbuf, ptr, mask, wsm, posdot, nslots, poolbf);
  k_gemm_hmem<<<dim3(H_DIM / 16, B_ROWS / 128), 256, 0, stream>>>(poolbf, WtS, bsp, hmem, Acat);
  k_gemm_lstm<<<dim3(H_DIM / 16, B_ROWS / 128), 256, 0, stream>>>(Acat, WtC, bih, cl, h_new, c_new);
}
